// BiLSTMBaseline_56049323213769
// MI455X (gfx1250) — matrix operand order repaired and checked by reading, not compiled, not run on hardware
//
#include <hip/hip_runtime.h>
#include <cmath>

// Shapes from the reference
#define Vv 32000
#define Ee 256      // embed dim
#define Hh 128      // hidden
#define Ll 40       // labels
#define Bb 4        // batch
#define Tt 512      // seq len
#define Gg 512      // 4H (gates)
#define Dd 256      // 2H (bi concat)

typedef __attribute__((ext_vector_type(16))) _Float16 v16h;
typedef __attribute__((ext_vector_type(8)))  float    v8f;
typedef __attribute__((ext_vector_type(4)))  float    v4f;   // native vector: OK for nontemporal builtins

__device__ __forceinline__ float sigf(float x) { return 1.0f / (1.0f + __expf(-x)); }
// robust tanh from one v_exp: 1 - 2/(e^{2x}+1)  (-> -1 as e->0, +1 as e->inf)
__device__ __forceinline__ float tanhfast(float x) {
    float e = __expf(2.0f * x);
    return 1.0f - 2.0f / (e + 1.0f);
}

// ---------------------------------------------------------------------------
// Kernel 1: embedding gather + input projection, both directions.
//   xp[dir][t][b][g] = sum_e embed[x[b,t], e] * Wih_dir[g, e] + b_dir[g]
// One wave per 16-row tile; 4 N-tiles per wave. kk-outer / nt-inner keeps
// 4 independent WMMA accumulator chains in flight and reuses the A fragment
// across consecutive v_wmma issues.
// ---------------------------------------------------------------------------
__global__ void __launch_bounds__(32)
k_xproj(const int* __restrict__ x, const float* __restrict__ embed,
        const float* __restrict__ Wih_f, const float* __restrict__ b_f,
        const float* __restrict__ Wih_b, const float* __restrict__ b_b,
        float* __restrict__ xp)
{
    const int lane  = threadIdx.x & 31;
    const int mtile = blockIdx.x;         // 0..127  (T*B/16 row tiles)
    const int ngrp  = blockIdx.y;         // 0..7    (4 col tiles each)
    const int dir   = blockIdx.z;         // 0..1
    const float* Wih  = dir ? Wih_b : Wih_f;
    const float* bias = dir ? b_b   : b_f;
    float* out = xp + (size_t)dir * ((size_t)Tt * Bb * Gg);

    const int row   = mtile * 16 + (lane & 15);   // flat row r = t*B + b
    const int t     = row >> 2;                   // B == 4
    const int b     = row & 3;
    const int tok   = x[b * Tt + t];
    const float* arow = embed + (size_t)tok * Ee;

    const int khalf = (lane >> 4) * 8;            // unused; A also uses kb16 (k = 16h + j)
    const int kb16  = (lane >> 4) * 16;           // B layout half-wave k base
    const int ncol  = lane & 15;
    const int mhalf = (lane >> 4) * 8;            // C/D layout m base

    // Preload all A fragments (K = 256 -> 8 steps of 32)
    v16h afrag[8];
#pragma unroll
    for (int kk = 0; kk < 8; ++kk) {
#pragma unroll
        for (int j = 0; j < 16; ++j) {
            const int k = kk * 32 + kb16 + j;
            afrag[kk][j] = (_Float16)arow[k];
        }
    }

    v8f acc[4] = {};
#pragma unroll
    for (int kk = 0; kk < 8; ++kk) {
        v16h bfrag[4];                     // B[k][n] = Wih[n][k]
#pragma unroll
        for (int nt = 0; nt < 4; ++nt) {
            const int n = (ngrp * 4 + nt) * 16 + ncol;
#pragma unroll
            for (int j = 0; j < 16; ++j)
                bfrag[nt][j] = (_Float16)Wih[(size_t)n * Ee + kk * 32 + kb16 + j];
        }
#pragma unroll
        for (int nt = 0; nt < 4; ++nt)
            acc[nt] = __builtin_amdgcn_wmma_f32_16x16x32_f16(
                false, afrag[kk], false, bfrag[nt], (short)0, acc[nt], false, false);
    }

#pragma unroll
    for (int nt = 0; nt < 4; ++nt) {
        const int n  = (ngrp * 4 + nt) * 16 + ncol;
        const float bn = bias[n];
#pragma unroll
        for (int i = 0; i < 8; ++i) {
            const int r = mtile * 16 + mhalf + i;
            out[(size_t)r * Gg + n] = acc[nt][i] + bn;
        }
    }
}

// ---------------------------------------------------------------------------
// Kernel 2: sequential LSTM recurrence, one persistent block per direction.
// Whh^T lives in registers as WMMA B-fragments (16 frags = 128 VGPRs/lane,
// 8 waves x 64 gate columns). h,c live in LDS. Next step's xp slab is
// prefetched (global_prefetch_b8) while this step's WMMAs run.
// ---------------------------------------------------------------------------
__global__ void __launch_bounds__(256)
k_recur(const float* __restrict__ xp,
        const float* __restrict__ Whh_f, const float* __restrict__ Whh_b,
        float* __restrict__ hs)
{
    extern __shared__ float smem[];
    float* g    = smem;              // [4][512] gate pre-activations
    float* hbuf = g + Bb * Gg;       // [4][128]
    float* cbuf = hbuf + Bb * Hh;    // [4][128]

    const int tid  = threadIdx.x;
    const int lane = tid & 31;
    const int wave = tid >> 5;       // 8 waves, 64 gate columns each
    const int dir  = blockIdx.x;
    const float* Whh = dir ? Whh_b : Whh_f;
    const float* xpd = xp + (size_t)dir * ((size_t)Tt * Bb * Gg);

    const int row   = lane & 15;
    const int khalf = (lane >> 4) * 8;
    const int kb16  = (lane >> 4) * 16;
    const int ncol  = lane & 15;
    const int mhalf = (lane >> 4) * 8;
    const int wn0   = wave * 64;

    // Register-resident Whh^T fragments: B[k][n] = Whh[n*H + k]
    v16h bf[4][4];
#pragma unroll
    for (int kk = 0; kk < 4; ++kk)
#pragma unroll
        for (int nt = 0; nt < 4; ++nt) {
            const int n = wn0 + nt * 16 + ncol;
#pragma unroll
            for (int j = 0; j < 16; ++j)
                bf[kk][nt][j] = (_Float16)Whh[(size_t)n * Hh + kk * 32 + kb16 + j];
        }

    for (int u = tid; u < Bb * Hh; u += 256) { hbuf[u] = 0.0f; cbuf[u] = 0.0f; }
    __syncthreads();

    for (int s = 0; s < Tt; ++s) {
        const int t = dir ? (Tt - 1 - s) : s;

        // prefetch next step's xp slab (8 KB) while WMMAs run
        if (s + 1 < Tt) {
            const int tn = dir ? (Tt - 2 - s) : (s + 1);
            __builtin_prefetch(xpd + (size_t)tn * Bb * Gg + tid * 8, 0, 0);
        }

        // A fragments from h (batch rows 0..3 valid, rest zero-padded)
        v16h af[4];
#pragma unroll
        for (int kk = 0; kk < 4; ++kk) {
#pragma unroll
            for (int j = 0; j < 16; ++j) {
                const int k = kk * 32 + kb16 + j;
                af[kk][j] = (row < Bb) ? (_Float16)hbuf[row * Hh + k]
                                       : (_Float16)0.0f;
            }
        }

        // kk-outer / nt-inner: 4 independent accumulator chains, A reuse
        v8f acc[4] = {};
#pragma unroll
        for (int kk = 0; kk < 4; ++kk)
#pragma unroll
            for (int nt = 0; nt < 4; ++nt)
                acc[nt] = __builtin_amdgcn_wmma_f32_16x16x32_f16(
                    false, af[kk], false, bf[kk][nt], (short)0, acc[nt], false, false);

        // g = (h @ Whh^T) + xp[t]
        const float* xpt = xpd + (size_t)t * Bb * Gg;
#pragma unroll
        for (int nt = 0; nt < 4; ++nt) {
            const int n = wn0 + nt * 16 + ncol;
#pragma unroll
            for (int i = 0; i < 8; ++i) {
                const int m = mhalf + i;
                if (m < Bb) g[m * Gg + n] = acc[nt][i] + xpt[m * Gg + n];
            }
        }
        __syncthreads();

        // gates (torch order i,f,g,o), update c/h, emit h
        for (int u = tid; u < Bb * Hh; u += 256) {
            const int b_ = u >> 7;
            const int k_ = u & (Hh - 1);
            const float gi = g[b_ * Gg + k_];
            const float gf = g[b_ * Gg + Hh + k_];
            const float gc = g[b_ * Gg + 2 * Hh + k_];
            const float go = g[b_ * Gg + 3 * Hh + k_];
            const float c  = sigf(gf) * cbuf[u] + sigf(gi) * tanhfast(gc);
            const float h  = sigf(go) * tanhfast(c);
            cbuf[u] = c;
            hbuf[u] = h;
            hs[((size_t)b_ * Tt + t) * Dd + dir * Hh + k_] = h;
        }
        __syncthreads();
    }
}

// ---------------------------------------------------------------------------
// Kernel 3: head projections (tiny, ~86 MFLOP): e_i/e_j/l_i/l_j per row.
// ---------------------------------------------------------------------------
__global__ void __launch_bounds__(256)
k_heads(const float* __restrict__ hs, const float* __restrict__ We,
        const float* __restrict__ Wl,
        float* __restrict__ e_i, float* __restrict__ e_j,
        float* __restrict__ l_i, float* __restrict__ l_j)
{
    const int NO  = 2 + 2 * Ll;     // 82 outputs per row
    const int gid = blockIdx.x * 256 + threadIdx.x;
    if (gid >= Bb * Tt * NO) return;
    const int r = gid / NO;
    const int o = gid - r * NO;

    const float* wrow;
    if (o == 0)           wrow = We;                                  // We[:256]
    else if (o == 1)      wrow = We + Dd;                             // We[256:]
    else if (o < 2 + Ll)  wrow = Wl + (size_t)(o - 2) * (2 * Dd);     // Wl[l,:256]
    else                  wrow = Wl + (size_t)(o - 2 - Ll) * (2 * Dd) + Dd;

    const float* hrow = hs + (size_t)r * Dd;
    float acc = 0.0f;
#pragma unroll 4
    for (int k = 0; k < Dd; ++k) acc += hrow[k] * wrow[k];

    if (o == 0)          e_i[r] = acc;
    else if (o == 1)     e_j[r] = acc;
    else if (o < 2 + Ll) l_i[(size_t)r * Ll + (o - 2)] = acc;
    else                 l_j[(size_t)r * Ll + (o - 2 - Ll)] = acc;
}

// ---------------------------------------------------------------------------
// Kernel 4: bandwidth-bound broadcast (168 MB label + 4 MB edge writes).
// One block per (b,i); l_i row + biases cached in LDS; fully contiguous
// 16-byte NON-TEMPORAL stores so the write-once stream doesn't evict the
// reused l_j rows (82 KB/batch, L2-resident) -- ~7.3 us HBM floor.
// ---------------------------------------------------------------------------
__global__ void __launch_bounds__(256)
k_bcast(const float* __restrict__ e_i, const float* __restrict__ e_j,
        const float* __restrict__ l_i, const float* __restrict__ l_j,
        const float* __restrict__ be, const float* __restrict__ bl,
        float* __restrict__ edge, float* __restrict__ label)
{
    const int i   = blockIdx.x;
    const int b   = blockIdx.y;
    const int tid = threadIdx.x;
    __shared__ float li_s[Ll];
    __shared__ float ei_s;
    if (tid < Ll) li_s[tid] = l_i[((size_t)b * Tt + i) * Ll + tid] + bl[tid];
    if (tid == 0) ei_s = e_i[b * Tt + i] + be[0];
    __syncthreads();

    float* erow = edge + (size_t)b * Tt * Tt + (size_t)i * Tt;
    for (int j = tid; j < Tt; j += 256)
        __builtin_nontemporal_store(ei_s + e_j[b * Tt + j], &erow[j]);

    float*      lrow = label + ((size_t)b * Tt * Tt + (size_t)i * Tt) * Ll;
    const v4f*  ljv  = (const v4f*)(l_j + (size_t)b * Tt * Ll);
    v4f*        outv = (v4f*)lrow;
    const int NC = Ll / 4;                       // 10 x 16B chunks per row
    for (int f = tid; f < Tt * NC; f += 256) {
        const int j = f / NC;
        const int q = f - j * NC;
        const v4f v = ljv[j * NC + q];
        v4f r;
        r.x = v.x + li_s[q * 4 + 0];
        r.y = v.y + li_s[q * 4 + 1];
        r.z = v.z + li_s[q * 4 + 2];
        r.w = v.w + li_s[q * 4 + 3];
        __builtin_nontemporal_store(r, &outv[f]);   // contiguous: j*40 + q*4
    }
}

// ---------------------------------------------------------------------------
extern "C" void kernel_launch(void* const* d_in, const int* in_sizes, int n_in,
                              void* d_out, int out_size, void* d_ws, size_t ws_size,
                              hipStream_t stream)
{
    (void)in_sizes; (void)n_in; (void)out_size; (void)ws_size;
    const int*   x     = (const int*)  d_in[0];
    const float* embed = (const float*)d_in[1];
    const float* Wih_f = (const float*)d_in[2];
    const float* Whh_f = (const float*)d_in[3];
    const float* b_f   = (const float*)d_in[4];
    const float* Wih_b = (const float*)d_in[5];
    const float* Whh_b = (const float*)d_in[6];
    const float* b_b   = (const float*)d_in[7];
    const float* We    = (const float*)d_in[8];
    const float* be    = (const float*)d_in[9];
    const float* Wl    = (const float*)d_in[10];
    const float* bl    = (const float*)d_in[11];

    // workspace carve (floats): xp | hs | e_i | e_j | l_i | l_j  (~11.2 MB)
    float* ws  = (float*)d_ws;
    float* xp  = ws;                                        // 2*T*B*4H
    float* hsb = xp  + (size_t)2 * Tt * Bb * Gg;            // B*T*2H
    float* e_i = hsb + (size_t)Bb * Tt * Dd;                // B*T
    float* e_j = e_i + (size_t)Bb * Tt;                     // B*T
    float* l_i = e_j + (size_t)Bb * Tt;                     // B*T*L
    float* l_j = l_i + (size_t)Bb * Tt * Ll;                // B*T*L

    float* edge  = (float*)d_out;                           // [B, T*T]
    float* label = edge + (size_t)Bb * Tt * Tt;             // [B, T*T, L]

    // 1) gather + input projection (WMMA)
    k_xproj<<<dim3(Tt * Bb / 16, Gg / 64, 2), 32, 0, stream>>>(
        x, embed, Wih_f, b_f, Wih_b, b_b, xp);

    // 2) recurrence: 2 persistent blocks (fwd/bwd), 12 KB dynamic LDS each
    const size_t lds = (size_t)(Bb * Gg + 2 * Bb * Hh) * sizeof(float);
    k_recur<<<2, 256, lds, stream>>>(xp, Whh_f, Whh_b, hsb);

    // 3) head projections
    const int nthr = Bb * Tt * (2 + 2 * Ll);
    k_heads<<<(nthr + 255) / 256, 256, 0, stream>>>(hsb, We, Wl, e_i, e_j, l_i, l_j);

    // 4) broadcast outer sums (HBM-bound, NT stores)
    k_bcast<<<dim3(Tt, Bb), 256, 0, stream>>>(e_i, e_j, l_i, l_j, be, bl, edge, label);
}
